// IGNNet_36601711296788
// MI455X (gfx1250) — matrix operand order repaired and checked by reading, not compiled, not run on hardware
//
#include <hip/hip_runtime.h>

#define BGRAPH 32
#define MNODES 128
#define DDIM   64
#define EPGC   1024
#define NNODES (BGRAPH * MNODES)
#define NEDGES (BGRAPH * EPGC)
#define C0PAD  160   // 129 channels padded to 5 k-chunks of 32

typedef _Float16 half_t;
typedef _Float16 v16h __attribute__((ext_vector_type(16)));
typedef _Float16 v8h  __attribute__((ext_vector_type(8)));
typedef float    v8f  __attribute__((ext_vector_type(8)));
typedef float    v16f __attribute__((ext_vector_type(16)));

// ---------------------------------------------------------------------------
// Build z0[b, i, j, c] = x[b, c, i, j] (layer view), f32, padded to 160 ch.
// Layer view x[b,c,i,j] = z_old[b, j, i, c]  (reference transpose (0,3,2,1)),
// so edge (src,dst) with features [1, emb_e] lands at zs[g, dst, src, 0..64].
// ---------------------------------------------------------------------------
__global__ void k_build_adj(const int* __restrict__ eidx, const int* __restrict__ etype,
                            const float* __restrict__ emb_e, float* __restrict__ z0) {
  int ed = blockIdx.x;
  int c  = threadIdx.x;                       // 0..63
  int srcg = eidx[ed];
  int dstg = eidx[NEDGES + ed];
  int g  = srcg / MNODES;
  int sl = srcg - g * MNODES;
  int dl = dstg - g * MNODES;
  size_t base = ((size_t)(g * MNODES + dl) * MNODES + sl) * C0PAD;
  atomicAdd(&z0[base + 1 + c], emb_e[(size_t)etype[ed] * DDIM + c]);
  if (c == 0) atomicAdd(&z0[base], 1.0f);
}

__global__ void k_set_diag(const int* __restrict__ h, const float* __restrict__ emb_h,
                           float* __restrict__ z0) {
  int n = blockIdx.x;                         // node 0..4095
  int c = threadIdx.x;                        // 0..63
  int b = n / MNODES, i = n % MNODES;
  size_t base = ((size_t)(b * MNODES + i) * MNODES + i) * C0PAD;
  z0[base + 65 + c] = emb_h[(size_t)h[n] * DDIM + c];   // diag node features
  if (c == 0) z0[base] += 1.0f;                          // adj[:,i,i,0] += 1 (after scatter)
}

// ---------------------------------------------------------------------------
// Pooled stats: diag[b,c,i], srow[b,c,i]=sum_j, scol[b,c,j]=sum_i  (stride Creal)
// ---------------------------------------------------------------------------
template <typename T>
__global__ void k_pool(const T* __restrict__ zin, int Cpad, int Creal,
                       float* __restrict__ dg, float* __restrict__ sr,
                       float* __restrict__ sc) {
  int b = blockIdx.x / MNODES, m = blockIdx.x % MNODES, c = threadIdx.x;
  if (c >= Creal) return;
  float r = 0.f, q = 0.f;
  const T* rowp = zin + ((size_t)(b * MNODES + m) * MNODES) * Cpad + c;
  const T* colp = zin + ((size_t)(b * MNODES) * MNODES + m) * Cpad + c;
  for (int j = 0; j < MNODES; ++j)  r += (float)rowp[(size_t)j * Cpad];
  for (int i = 0; i < MNODES; ++i)  q += (float)colp[(size_t)i * MNODES * Cpad];
  size_t o = ((size_t)b * Creal + c) * MNODES + m;
  sr[o] = r;
  sc[o] = q;
  dg[o] = (float)zin[((size_t)(b * MNODES + m) * MNODES + m) * Cpad + c];
}

__global__ void k_sds(const float* __restrict__ dg, const float* __restrict__ sr,
                      int Creal, float* __restrict__ sd, float* __restrict__ sa) {
  int b = blockIdx.x, c = threadIdx.x;
  if (c >= Creal) return;
  float d = 0.f, a = 0.f;
  const float* dp = dg + ((size_t)b * Creal + c) * MNODES;
  const float* sp = sr + ((size_t)b * Creal + c) * MNODES;
  for (int i = 0; i < MNODES; ++i) { d += dp[i]; a += sp[i]; }
  sd[b * Creal + c] = d;
  sa[b * Creal + c] = a;
}

// ---------------------------------------------------------------------------
// Broadcast vectors (basis ops 1-9, 12-15 + biases) for the GEMM epilogue.
// coeffs layout: [d_in][64][15]
// ---------------------------------------------------------------------------
__global__ void k_poolvec(const float* __restrict__ dg, const float* __restrict__ sr,
                          const float* __restrict__ sc, const float* __restrict__ sd,
                          const float* __restrict__ sa, int Creal,
                          const float* __restrict__ coeffs,
                          const float* __restrict__ db, const float* __restrict__ ab,
                          float* __restrict__ rv, float* __restrict__ cv,
                          float* __restrict__ dv, float* __restrict__ kv) {
  int b = blockIdx.x / MNODES, i = blockIdx.x % MNODES, s = threadIdx.x;
  const float im = 1.0f / MNODES, im2 = im * im;
  float arv = 0.f, acv = 0.f, adv = 0.f, akv = 0.f;
  for (int d = 0; d < Creal; ++d) {
    size_t o = ((size_t)b * Creal + d) * MNODES + i;
    float fd = dg[o], fr = sr[o], fc = sc[o];
    float fsd = sd[b * Creal + d], fsa = sa[b * Creal + d];
    const float* c = coeffs + ((size_t)d * DDIM + s) * 15;
    adv += fd * c[0] + (fr * c[2] + fc * c[3]) * im + fsd * c[1] * im + fsa * c[4] * im2;
    arv += (fc * c[5] + fr * c[6]) * im + fd * c[11];
    acv += (fc * c[7] + fr * c[8]) * im + fd * c[12];
    akv += fsd * c[13] * im + fsa * c[14] * im2;
  }
  size_t o = ((size_t)b * DDIM + s) * MNODES + i;
  rv[o] = arv;
  cv[o] = acv;
  dv[o] = adv + db[s];
  if (i == 0) kv[b * DDIM + s] = akv + ab[s];
}

// ---------------------------------------------------------------------------
// Swizzle coeff ops 9/10 into the per-lane WMMA A-operand layout (f16).
// Wsw[(mat*4+st)*KC + kc][lane][e] with A k-map: k = 16*half+e
// ---------------------------------------------------------------------------
__global__ void k_swz(const float* __restrict__ coeffs, half_t* __restrict__ Wsw,
                      int Creal, int KC) {
  int t = threadIdx.x;                 // 0..511
  int l = t >> 4, e = t & 15;
  int lh = l >> 4, l16 = l & 15;
  int kc = blockIdx.x % KC;
  int stm = blockIdx.x / KC;
  int st = stm & 3, mat = stm >> 2;
  int kloc = 16 * lh + e;
  int d = kc * 32 + kloc;
  int s = st * 16 + l16;
  int op = mat ? 10 : 9;
  float v = (d < Creal) ? coeffs[((size_t)d * DDIM + s) * 15 + op] : 0.f;
  Wsw[(size_t)blockIdx.x * 512 + t] = (_Float16)v;
}

// ---------------------------------------------------------------------------
// Main fused 2->2 layer: WMMA channel-mix GEMM (ops 10,11) + full epilogue + ReLU.
// One wave per (b, i, 16-j group); 4 s-tiles; D = 16s x 16pos, f32 accum.
// ---------------------------------------------------------------------------
template <typename TIn, int KC>
__global__ void k_gemm22(const TIn* __restrict__ zin, const half_t* __restrict__ Wsw,
                         const float* __restrict__ rv, const float* __restrict__ cv,
                         const float* __restrict__ dv, const float* __restrict__ kv,
                         half_t* __restrict__ zout) {
  const int Cpad = KC * 32;
  unsigned wid = (blockIdx.x * blockDim.x + threadIdx.x) >> 5;
  int lane = threadIdx.x & 31;
  int lh = lane >> 4, l16 = lane & 15;
  int jt = wid & 7;
  int i  = (wid >> 3) & 127;
  int b  = wid >> 10;
  int p  = jt * 16 + l16;              // this lane's output column j

  v8f acc[4] = {};
  const TIn* bij_base = zin + ((size_t)((b * MNODES + i) * MNODES) + p) * Cpad + 16 * lh;
  const TIn* bt_base  = zin + ((size_t)((b * MNODES + p) * MNODES) + i) * Cpad + 16 * lh;

#pragma unroll
  for (int kc = 0; kc < KC; ++kc) {
    v16h bij, bt;
    if constexpr (sizeof(TIn) == 4) {
      v16f f0 = *(const v16f*)(bij_base + kc * 32);
      v16f f1 = *(const v16f*)(bt_base + kc * 32);
#pragma unroll
      for (int ee = 0; ee < 16; ++ee) {
        bij[ee] = (_Float16)f0[ee];
        bt[ee]  = (_Float16)f1[ee];
      }
    } else {
      bij = *(const v16h*)(bij_base + kc * 32);
      bt  = *(const v16h*)(bt_base + kc * 32);
    }
#pragma unroll
    for (int st = 0; st < 4; ++st) {
      v16h a9  = *(const v16h*)(Wsw + (size_t)(((0 * 4 + st) * KC + kc) * 512) + lane * 16);
      acc[st] = __builtin_amdgcn_wmma_f32_16x16x32_f16(false, a9, false, bij,
                                                       (short)0, acc[st], false, false);
      v16h a10 = *(const v16h*)(Wsw + (size_t)(((1 * 4 + st) * KC + kc) * 512) + lane * 16);
      acc[st] = __builtin_amdgcn_wmma_f32_16x16x32_f16(false, a10, false, bt,
                                                       (short)0, acc[st], false, false);
    }
  }

  // epilogue: + row_vec[s,i] + col_vec[s,j] + const[s] (+ diag_vec on i==j), ReLU, f16 store
#pragma unroll
  for (int st = 0; st < 4; ++st) {
    v8h hv;
#pragma unroll
    for (int r = 0; r < 8; ++r) {
      int s = st * 16 + 8 * lh + r;
      size_t so = ((size_t)b * DDIM + s) * MNODES;
      float v = acc[st][r] + rv[so + i] + cv[so + p] + kv[b * DDIM + s];
      if (p == i) v += dv[so + i];
      v = v > 0.f ? v : 0.f;
      hv[r] = (_Float16)v;
    }
    *(v8h*)(zout + ((size_t)((b * MNODES + i) * MNODES) + p) * DDIM + st * 16 + 8 * lh) = hv;
  }
}

// ---------------------------------------------------------------------------
// 2->1 layer + mean_nodes + MLP readout, collapsed to sdiag/sall algebra.
// hg[b,s] = (1/m) * sum_d [ sdiag*(c0+c1) + sall*(c2+c3+c4)/m ] + b21[s]
// ---------------------------------------------------------------------------
__global__ void k_head(const float* __restrict__ sd, const float* __restrict__ sa,
                       const float* __restrict__ c21, const float* __restrict__ b21,
                       const float* __restrict__ w1, const float* __restrict__ b1,
                       const float* __restrict__ w2, const float* __restrict__ b2,
                       const float* __restrict__ w3, const float* __restrict__ b3,
                       float* __restrict__ out) {
  __shared__ float hg[64], y1[32], y2[16];
  int b = blockIdx.x, s = threadIdx.x;
  const float im = 1.0f / MNODES;
  float a = 0.f;
  for (int d = 0; d < DDIM; ++d) {
    const float* c = c21 + ((size_t)d * DDIM + s) * 5;
    a += sd[b * DDIM + d] * (c[0] + c[1]) + sa[b * DDIM + d] * (c[2] + c[3] + c[4]) * im;
  }
  hg[s] = a * im + b21[s];
  __syncthreads();
  if (s < 32) {
    float y = b1[s];
    for (int k = 0; k < 64; ++k) y += hg[k] * w1[k * 32 + s];
    y1[s] = y > 0.f ? y : 0.f;
  }
  __syncthreads();
  if (s < 16) {
    float y = b2[s];
    for (int k = 0; k < 32; ++k) y += y1[k] * w2[k * 16 + s];
    y2[s] = y > 0.f ? y : 0.f;
  }
  __syncthreads();
  if (s == 0) {
    float y = b3[0];
    for (int k = 0; k < 16; ++k) y += y2[k] * w3[k];
    out[b] = y;
  }
}

// ---------------------------------------------------------------------------
extern "C" void kernel_launch(void* const* d_in, const int* in_sizes, int n_in,
                              void* d_out, int out_size, void* d_ws, size_t ws_size,
                              hipStream_t stream) {
  (void)in_sizes; (void)n_in; (void)out_size; (void)ws_size;

  const int*   h     = (const int*)d_in[0];
  const int*   et    = (const int*)d_in[1];
  const int*   eidx  = (const int*)d_in[2];
  // d_in[3] batch_info implied by src/M
  const float* emb_h = (const float*)d_in[4];
  const float* emb_e = (const float*)d_in[5];
  const float* c22[3] = {(const float*)d_in[6], (const float*)d_in[9], (const float*)d_in[12]};
  const float* db_[3] = {(const float*)d_in[7], (const float*)d_in[10], (const float*)d_in[13]};
  const float* ab_[3] = {(const float*)d_in[8], (const float*)d_in[11], (const float*)d_in[14]};
  const float* c21 = (const float*)d_in[15];
  const float* b21 = (const float*)d_in[16];
  const float* w1 = (const float*)d_in[17];
  const float* b1 = (const float*)d_in[18];
  const float* w2 = (const float*)d_in[19];
  const float* b2 = (const float*)d_in[20];
  const float* w3 = (const float*)d_in[21];
  const float* b3 = (const float*)d_in[22];

  char* ws = (char*)d_ws;
  size_t off = 0;
  auto alloc = [&](size_t bytes) -> void* {
    off = (off + 255) & ~(size_t)255;
    void* p = ws + off;
    off += bytes;
    return p;
  };

  const size_t z0_bytes = (size_t)BGRAPH * MNODES * MNODES * C0PAD * 4;
  float*  z0  = (float*)alloc(z0_bytes);
  half_t* z1  = (half_t*)alloc((size_t)BGRAPH * MNODES * MNODES * DDIM * 2);
  half_t* z2  = (half_t*)alloc((size_t)BGRAPH * MNODES * MNODES * DDIM * 2);
  float*  pdg = (float*)alloc((size_t)BGRAPH * C0PAD * MNODES * 4);
  float*  psr = (float*)alloc((size_t)BGRAPH * C0PAD * MNODES * 4);
  float*  psc = (float*)alloc((size_t)BGRAPH * C0PAD * MNODES * 4);
  float*  sd  = (float*)alloc((size_t)BGRAPH * C0PAD * 4);
  float*  sa  = (float*)alloc((size_t)BGRAPH * C0PAD * 4);
  float*  rv  = (float*)alloc((size_t)BGRAPH * DDIM * MNODES * 4);
  float*  cv  = (float*)alloc((size_t)BGRAPH * DDIM * MNODES * 4);
  float*  dv  = (float*)alloc((size_t)BGRAPH * DDIM * MNODES * 4);
  float*  kv  = (float*)alloc((size_t)BGRAPH * DDIM * 4);
  half_t* Wsw = (half_t*)alloc((size_t)8 * 5 * 512 * 2);

  const int gemm_blocks = (BGRAPH * MNODES * (MNODES / 16)) / 8;  // 8 waves / block

  // ---- build dense input tensor z0 (f32, padded 160ch) ----
  hipMemsetAsync(z0, 0, z0_bytes, stream);
  k_build_adj<<<NEDGES, 64, 0, stream>>>(eidx, et, emb_e, z0);
  k_set_diag<<<NNODES, 64, 0, stream>>>(h, emb_h, z0);

  // ---- layer 0: f32 in (C=129 pad 160), f16 out ----
  k_pool<float><<<BGRAPH * MNODES, 160, 0, stream>>>(z0, C0PAD, 129, pdg, psr, psc);
  k_sds<<<BGRAPH, 160, 0, stream>>>(pdg, psr, 129, sd, sa);
  k_poolvec<<<BGRAPH * MNODES, 64, 0, stream>>>(pdg, psr, psc, sd, sa, 129,
                                                c22[0], db_[0], ab_[0], rv, cv, dv, kv);
  k_swz<<<8 * 5, 512, 0, stream>>>(c22[0], Wsw, 129, 5);
  k_gemm22<float, 5><<<gemm_blocks, 256, 0, stream>>>(z0, Wsw, rv, cv, dv, kv, z1);

  // ---- layer 1: f16 -> f16 ----
  k_pool<half_t><<<BGRAPH * MNODES, 64, 0, stream>>>(z1, DDIM, DDIM, pdg, psr, psc);
  k_sds<<<BGRAPH, 64, 0, stream>>>(pdg, psr, DDIM, sd, sa);
  k_poolvec<<<BGRAPH * MNODES, 64, 0, stream>>>(pdg, psr, psc, sd, sa, DDIM,
                                                c22[1], db_[1], ab_[1], rv, cv, dv, kv);
  k_swz<<<8 * 2, 512, 0, stream>>>(c22[1], Wsw, DDIM, 2);
  k_gemm22<half_t, 2><<<gemm_blocks, 256, 0, stream>>>(z1, Wsw, rv, cv, dv, kv, z2);

  // ---- layer 2: f16 -> f16 (reuse z1 buffer) ----
  k_pool<half_t><<<BGRAPH * MNODES, 64, 0, stream>>>(z2, DDIM, DDIM, pdg, psr, psc);
  k_sds<<<BGRAPH, 64, 0, stream>>>(pdg, psr, DDIM, sd, sa);
  k_poolvec<<<BGRAPH * MNODES, 64, 0, stream>>>(pdg, psr, psc, sd, sa, DDIM,
                                                c22[2], db_[2], ab_[2], rv, cv, dv, kv);
  k_swz<<<8 * 2, 512, 0, stream>>>(c22[2], Wsw, DDIM, 2);
  k_gemm22<half_t, 2><<<gemm_blocks, 256, 0, stream>>>(z2, Wsw, rv, cv, dv, kv, z1);

  // ---- 2->1 + mean + MLP head (needs only sdiag/sall of z3) ----
  k_pool<half_t><<<BGRAPH * MNODES, 64, 0, stream>>>(z1, DDIM, DDIM, pdg, psr, psc);
  k_sds<<<BGRAPH, 64, 0, stream>>>(pdg, psr, DDIM, sd, sa);
  k_head<<<BGRAPH, 64, 0, stream>>>(sd, sa, c21, b21, w1, b1, w2, b2, w3, b3,
                                    (float*)d_out);
}
